// GCN_23888608101019
// MI455X (gfx1250) — matrix operand order repaired and checked by reading, not compiled, not run on hardware
//
#include <hip/hip_runtime.h>
#include <hip/hip_bf16.h>

typedef __attribute__((ext_vector_type(16))) _Float16 v16h;
typedef __attribute__((ext_vector_type(8)))  float    v8f;

// Match the builtin's parameter type exactly: vector_size(16) int, AS1/AS3.
typedef int v4i_t __attribute__((vector_size(16)));
typedef v4i_t __attribute__((address_space(1)))* gv4i_ptr;
typedef v4i_t __attribute__((address_space(3)))* lv4i_ptr;

#define HID 128

// ---------------------------------------------------------------------------
// A-fragment loader: 16-bit A matrix 16x32 layout (ISA 7.12.2).
// lane (0..31): row M = lane&15, hi = lane>>4.
//   a[0..7]  = A[row][k0 + 8*hi + 0..7]
//   a[8..15] = A[row][k0 + 16 + 8*hi + 0..7]
// A is f32 row-major with stride HID; converted to f16 on the fly.
// ---------------------------------------------------------------------------
__device__ __forceinline__ v16h load_a_frag(const float* __restrict__ A,
                                            int row, int k0, int hi) {
  const float* p = A + (size_t)row * HID + k0 + 8 * hi;
  float4 f0 = *(const float4*)(p);
  float4 f1 = *(const float4*)(p + 4);
  float4 f2 = *(const float4*)(p + 16);
  float4 f3 = *(const float4*)(p + 20);
  v16h a;
  a[0]  = (_Float16)f0.x; a[1]  = (_Float16)f0.y;
  a[2]  = (_Float16)f0.z; a[3]  = (_Float16)f0.w;
  a[4]  = (_Float16)f1.x; a[5]  = (_Float16)f1.y;
  a[6]  = (_Float16)f1.z; a[7]  = (_Float16)f1.w;
  a[8]  = (_Float16)f2.x; a[9]  = (_Float16)f2.y;
  a[10] = (_Float16)f2.z; a[11] = (_Float16)f2.w;
  a[12] = (_Float16)f3.x; a[13] = (_Float16)f3.y;
  a[14] = (_Float16)f3.z; a[15] = (_Float16)f3.w;
  return a;
}

// ---------------------------------------------------------------------------
// Stage packed weights (4096*NT bytes) global -> LDS, once per workgroup.
// Uses gfx1250 GLOBAL_LOAD_ASYNC_TO_LDS_B128 (ASYNCcnt) when the builtin is
// available; plain copies otherwise. 256 threads, NT 16B chunks per thread.
// ---------------------------------------------------------------------------
template <int NT>
__device__ __forceinline__ void stage_b(const _Float16* __restrict__ Bp,
                                        _Float16* smem) {
  const int tid = threadIdx.x;
#if __has_builtin(__builtin_amdgcn_global_load_async_to_lds_b128)
#pragma unroll
  for (int i = 0; i < NT; ++i) {
    const size_t off = (size_t)(tid + i * 256) * 16;  // byte offset, 16B chunks
    __builtin_amdgcn_global_load_async_to_lds_b128(
        (gv4i_ptr)((const char*)Bp + off),
        (lv4i_ptr)((char*)smem + off),
        0, 0);
  }
#if __has_builtin(__builtin_amdgcn_s_wait_asynccnt)
  __builtin_amdgcn_s_wait_asynccnt(0);
#else
  asm volatile("s_wait_asynccnt 0" ::: "memory");
#endif
#else
#pragma unroll
  for (int i = 0; i < NT; ++i) {
    const int off = tid + i * 256;
    ((uint4*)smem)[off] = ((const uint4*)Bp)[off];
  }
#endif
  __syncthreads();
}

// ---------------------------------------------------------------------------
// Core WMMA strip: one wave computes rows [rowBase, rowBase+16) x NT*16 cols,
// K = 128 in 4 steps of v_wmma_f32_16x16x32_f16. B fragments come from LDS
// (ds_load_b128), each lane's fragment is 16 contiguous f16 (32 B).
// ---------------------------------------------------------------------------
template <int NT>
__device__ __forceinline__ void wmma_strip(const float* __restrict__ A,
                                           const _Float16* smem,
                                           int rowBase, int nrows, int lane,
                                           v8f acc[NT]) {
  const int hi = lane >> 4;
  int row = rowBase + (lane & 15);
  if (row >= nrows) row = nrows - 1;  // clamp: pad rows read valid memory
#pragma unroll
  for (int ki = 0; ki < 4; ++ki) {
    v16h a = load_a_frag(A, row, ki * 32, hi);
#pragma unroll
    for (int n = 0; n < NT; ++n) {
      const v16h b = *(const v16h*)(smem + ((size_t)(ki * NT + n) * 32 + lane) * 16);
      acc[n] = __builtin_amdgcn_wmma_f32_16x16x32_f16(
          false, a, false, b, (short)0, acc[n], false, false);
    }
  }
}

// Out[N x NT*16] = A[N x 128] @ W (packed) + bias (bias may be null)
template <int NT>
__global__ void gemm_bias_kernel(const float* __restrict__ A,
                                 const _Float16* __restrict__ Bp,
                                 const float* __restrict__ bias,
                                 float* __restrict__ Out, int nrows) {
  __shared__ _Float16 smem[2048 * NT];
  stage_b<NT>(Bp, smem);
  const int lane = threadIdx.x & 31;
  const int wave = threadIdx.x >> 5;
  const int rowBase = blockIdx.x * 128 + wave * 16;
  v8f z = {0.f, 0.f, 0.f, 0.f, 0.f, 0.f, 0.f, 0.f};
  v8f acc[NT];
#pragma unroll
  for (int n = 0; n < NT; ++n) acc[n] = z;
  wmma_strip<NT>(A, smem, rowBase, nrows, lane, acc);
  const int hi = lane >> 4;
  const int cn = lane & 15;
  const int ncols = NT * 16;
#pragma unroll
  for (int n = 0; n < NT; ++n) {
    const int col = n * 16 + cn;
    const float bv = bias ? bias[col] : 0.0f;
#pragma unroll
    for (int r = 0; r < 8; ++r) {
      const int row = rowBase + r + 8 * hi;  // C/D layout: VGPR r -> M=r(+8)
      if (row < nrows) Out[(size_t)row * ncols + col] = acc[n][r] + bv;
    }
  }
}

// C = relu(C + convB + H@linW + linB) + H   (new h written in place of C)
__global__ void gemm_lin_merge_kernel(const float* __restrict__ H,
                                      const _Float16* __restrict__ Bp,
                                      const float* __restrict__ linB,
                                      const float* __restrict__ convB,
                                      float* __restrict__ C, int nrows) {
  __shared__ _Float16 smem[2048 * 8];
  stage_b<8>(Bp, smem);
  const int lane = threadIdx.x & 31;
  const int wave = threadIdx.x >> 5;
  const int rowBase = blockIdx.x * 128 + wave * 16;
  v8f z = {0.f, 0.f, 0.f, 0.f, 0.f, 0.f, 0.f, 0.f};
  v8f acc[8];
#pragma unroll
  for (int n = 0; n < 8; ++n) acc[n] = z;
  wmma_strip<8>(H, smem, rowBase, nrows, lane, acc);
  const int hi = lane >> 4;
  const int cn = lane & 15;
#pragma unroll
  for (int n = 0; n < 8; ++n) {
    const int col = n * 16 + cn;
    const float lb = linB[col];
    const float cb = convB[col];
#pragma unroll
    for (int r = 0; r < 8; ++r) {
      const int row = rowBase + r + 8 * hi;
      if (row < nrows) {
        const size_t idx = (size_t)row * HID + col;
        float v = C[idx] + cb + acc[n][r] + lb;
        C[idx] = fmaxf(v, 0.0f) + H[idx];
      }
    }
  }
}

// ---------------------------------------------------------------------------
// Weight pre-pack: f32 W[128 x ncols] -> f16 fragments in B layout.
// P[((ki*ntiles + n)*32 + lane)*16 + t] = W[ki*32 + 8*(lane>>4) + (t&7) + 16*(t>>3)][n*16 + (lane&15)]
// ---------------------------------------------------------------------------
__global__ void pack_weight_kernel(const float* __restrict__ W,
                                   _Float16* __restrict__ P, int ncols) {
  const int p = blockIdx.x * 256 + threadIdx.x;
  const int total = 128 * ncols;
  if (p >= total) return;
  const int t = p & 15;
  const int lane = (p >> 4) & 31;
  const int rest = p >> 9;
  const int ntiles = ncols >> 4;
  const int n = rest % ntiles;
  const int ki = rest / ntiles;
  const int k = ki * 32 + 8 * (lane >> 4) + (t & 7) + 16 * (t >> 3);
  const int c = n * 16 + (lane & 15);
  P[p] = (_Float16)W[(size_t)k * ncols + c];
}

// ---------------------------------------------------------------------------
// Normalized-adjacency prep
// ---------------------------------------------------------------------------
__global__ void deg_init_kernel(float* __restrict__ deg, int n) {
  const int i = blockIdx.x * 256 + threadIdx.x;
  if (i < n) deg[i] = 1.0f;  // self loop
}

__global__ void deg_edge_kernel(const int* __restrict__ ei, float* __restrict__ deg,
                                int nE) {
  const int e = blockIdx.x * 256 + threadIdx.x;
  if (e >= nE) return;
  const int r = ei[e];
  const int c = ei[nE + e];
  if (r != c) atomicAdd(&deg[r], 1.0f);
}

__global__ void dinv_kernel(float* __restrict__ deg, int n) {
  const int i = blockIdx.x * 256 + threadIdx.x;
  if (i < n) {
    const float d = deg[i];
    deg[i] = (d > 0.0f) ? rsqrtf(d) : 0.0f;
  }
}

// C[i] = dinv[i]^2 * S[i]  (self-loop term; also initializes C)
__global__ void selfloop_kernel(const float* __restrict__ S,
                                const float* __restrict__ dinv,
                                float* __restrict__ C, int n) {
  const int tid = blockIdx.x * 256 + threadIdx.x;
  const int node = tid >> 5;
  const int part = tid & 31;
  if (node >= n) return;
  const float w = dinv[node] * dinv[node];
  const size_t idx = (size_t)node * HID + part * 4;
  float4 s = *(const float4*)(S + idx);
  float4 o = {w * s.x, w * s.y, w * s.z, w * s.w};
  *(float4*)(C + idx) = o;
}

// C[r] += dinv[r]*dinv[c] * S[c] for every edge (one wave per edge)
__global__ void edge_scatter_kernel(const int* __restrict__ ei,
                                    const float* __restrict__ dinv,
                                    const float* __restrict__ S,
                                    float* __restrict__ C, int nE) {
  const int e = blockIdx.x * 8 + (threadIdx.x >> 5);
  const int lane = threadIdx.x & 31;
  if (e >= nE) return;
  const int r = ei[e];
  const int c = ei[nE + e];
  if (r == c) return;  // original self-loops carry weight 0
  float* dst = C + (size_t)r * HID + lane * 4;
  __builtin_prefetch(dst, 1, 0);  // global_prefetch_b8 the atomic target line
  const float w = dinv[r] * dinv[c];
  float4 s = *(const float4*)(S + (size_t)c * HID + lane * 4);
  atomicAdd(dst + 0, w * s.x);
  atomicAdd(dst + 1, w * s.y);
  atomicAdd(dst + 2, w * s.z);
  atomicAdd(dst + 3, w * s.w);
}

// ---------------------------------------------------------------------------
extern "C" void kernel_launch(void* const* d_in, const int* in_sizes, int n_in,
                              void* d_out, int out_size, void* d_ws, size_t ws_size,
                              hipStream_t stream) {
  const float* x     = (const float*)d_in[0];
  const int*   ei    = (const int*)d_in[1];
  const float* W0    = (const float*)d_in[2];
  const float* b0    = (const float*)d_in[3];
  const float* convW = (const float*)d_in[4];
  const float* convB = (const float*)d_in[5];
  const float* linW  = (const float*)d_in[6];
  const float* linB  = (const float*)d_in[7];
  const float* Wf    = (const float*)d_in[8];
  const float* bf    = (const float*)d_in[9];
  float* out = (float*)d_out;

  const int nN = in_sizes[0] / HID;  // 50000
  const int nE = in_sizes[1] / 2;    // 800000

  // workspace carve-out (256B aligned)
  char* ws = (char*)d_ws;
  size_t off = 0;
  auto carve = [&](size_t bytes) {
    void* p = ws + off;
    off = (off + bytes + 255) & ~(size_t)255;
    return p;
  };
  float* deg  = (float*)carve((size_t)nN * 4);          // deg -> dinv in place
  float* bufA = (float*)carve((size_t)nN * HID * 4);    // h
  float* bufB = (float*)carve((size_t)nN * HID * 4);    // support
  float* bufC = (float*)carve((size_t)nN * HID * 4);    // nei / next h
  _Float16* W0p    = (_Float16*)carve(128 * 128 * 2);
  _Float16* convWp = (_Float16*)carve(3 * 128 * 128 * 2);
  _Float16* linWp  = (_Float16*)carve(3 * 128 * 128 * 2);
  _Float16* Wfp    = (_Float16*)carve(128 * 64 * 2);

  // --- pack all weights into WMMA B-fragment order ---
  pack_weight_kernel<<<64, 256, 0, stream>>>(W0, W0p, 128);
  for (int i = 0; i < 3; ++i) {
    pack_weight_kernel<<<64, 256, 0, stream>>>(convW + i * 128 * 128,
                                               convWp + i * 128 * 128, 128);
    pack_weight_kernel<<<64, 256, 0, stream>>>(linW + i * 128 * 128,
                                               linWp + i * 128 * 128, 128);
  }
  pack_weight_kernel<<<32, 256, 0, stream>>>(Wf, Wfp, 64);

  // --- degree / D^-1/2 ---
  const int gN = (nN + 255) / 256;
  deg_init_kernel<<<gN, 256, 0, stream>>>(deg, nN);
  deg_edge_kernel<<<(nE + 255) / 256, 256, 0, stream>>>(ei, deg, nE);
  dinv_kernel<<<gN, 256, 0, stream>>>(deg, nN);

  const int g128 = (nN + 127) / 128;

  // h = x @ W0 + b0
  gemm_bias_kernel<8><<<g128, 256, 0, stream>>>(x, W0p, b0, bufA, nN);

  float* A = bufA;
  float* C = bufC;
  for (int i = 0; i < 3; ++i) {
    // support = h @ convW[i]
    gemm_bias_kernel<8><<<g128, 256, 0, stream>>>(A, convWp + i * 128 * 128,
                                                  nullptr, bufB, nN);
    // nei = D^-1/2 A D^-1/2 @ support  (self-loop init + edge scatter)
    selfloop_kernel<<<((size_t)nN * 32 + 255) / 256, 256, 0, stream>>>(bufB, deg, C, nN);
    edge_scatter_kernel<<<(nE + 7) / 8, 256, 0, stream>>>(ei, deg, bufB, C, nE);
    // h' = relu(nei + convB + h@linW + linB) + h   (in place of C)
    gemm_lin_merge_kernel<<<g128, 256, 0, stream>>>(A, linWp + i * 128 * 128,
                                                    linB + i * 128, convB + i * 128,
                                                    C, nN);
    float* t = A; A = C; C = t;  // rotate: new h is in old C
  }

  // out = h @ Wf + bf  (64 output cols -> 4 column tiles)
  gemm_bias_kernel<4><<<g128, 256, 0, stream>>>(A, Wfp, bf, out, nN);
}
